// MultiHeadAttention_60120952209427
// MI455X (gfx1250) — matrix operand order repaired and checked by reading, not compiled, not run on hardware
//
#include <hip/hip_runtime.h>
#include <hip/hip_bf16.h>
#include <cstdint>

#define DMODEL 1024
#define NHEADS 16
#define DK     64
#define SEQ    2048
#define BATCH  2
#define MROWS  (BATCH*SEQ)   // 4096
#define EPSR   1e-8f

typedef __bf16 bf16_t;
typedef unsigned int u32;
typedef unsigned long long u64;
typedef __attribute__((ext_vector_type(16))) __bf16 v16bf;
typedef __attribute__((ext_vector_type(8)))  __bf16 v8bf;
typedef __attribute__((ext_vector_type(8)))  float  v8f;
typedef __attribute__((ext_vector_type(4)))  u32    v4u;
typedef __attribute__((ext_vector_type(8)))  int    v8i;
typedef __attribute__((ext_vector_type(4)))  int    v4i;

#if defined(__has_builtin)
#if __has_builtin(__builtin_amdgcn_tensor_load_to_lds) && \
    __has_builtin(__builtin_amdgcn_s_wait_tensorcnt)
#define USE_TDM 1
#endif
#endif

#ifdef USE_TDM
// 2-D TDM load: nrows rows of row_len bf16 elements (global row stride
// row_stride elements) -> contiguous LDS tile. One TDM op per call.
// This toolchain's builtin is the 6-arg form:
//   (uint32x4 g0, int32x8 g1, int32x4 g2, int32x4 g3, int32x8 pad, i32 cpol)
static __device__ __forceinline__ void tdm_load_2d(const bf16_t* gsrc, void* lds_dst,
                                                   u32 row_len, u32 nrows,
                                                   u32 row_stride) {
  u64 ga = (u64)(uintptr_t)gsrc;
  v4u g0;
  g0.x = 1u;                               // count=1, user descriptor
  g0.y = (u32)(uintptr_t)lds_dst;          // lds_addr[31:0] (flat addr low bits)
  g0.z = (u32)ga;                          // global_addr[31:0]
  g0.w = (u32)(ga >> 32) | (2u << 30);     // global_addr[56:32], type=2 (image)
  u32 dim0 = row_len;                      // tensor_dim0
  u32 dim1 = 0x000FFFFFu;                  // tensor_dim1: large, no OOB clip
  v8i g1;
  g1[0] = (int)(1u << 16);                 // wg_mask=0, data_size=1 (2 bytes)
  g1[1] = (int)((dim0 & 0xFFFFu) << 16);   // atomic_barrier=0, tensor_dim0 lo16
  g1[2] = (int)(((dim0 >> 16) & 0xFFFFu) | ((dim1 & 0xFFFFu) << 16));
  g1[3] = (int)(((dim1 >> 16) & 0xFFFFu) | ((row_len & 0xFFFFu) << 16)); // tile_dim0
  g1[4] = (int)(nrows & 0xFFFFu);          // tile_dim1, tile_dim2=0
  g1[5] = (int)row_stride;                 // tensor_dim0_stride lo32
  g1[6] = 0;
  g1[7] = 0;
  v4i z4 = {};
  v8i z8 = {};
  __builtin_amdgcn_tensor_load_to_lds(g0, g1, z4, z4, z8, 0);
}
#define TDM_WAIT0() __builtin_amdgcn_s_wait_tensorcnt(0)
#else
// Fallback: wave-cooperative synchronous copy (issuing wave only).
static __device__ __forceinline__ void tdm_load_2d(const bf16_t* gsrc, void* lds_dst,
                                                   u32 row_len, u32 nrows,
                                                   u32 row_stride) {
  bf16_t* d = (bf16_t*)lds_dst;
  int lane = threadIdx.x & 31;
  for (u32 r = 0; r < nrows; ++r)
    for (u32 c = (u32)lane * 8; c < row_len; c += 32 * 8)
      *(v8bf*)(d + (size_t)r * row_len + c) =
          *(const v8bf*)(gsrc + (size_t)r * row_stride + c);
}
#define TDM_WAIT0() ((void)0)
#endif

static __device__ __forceinline__ v8f wmma_bf16(v16bf a, v16bf b, v8f c) {
  // D(16x16 f32) = A(16x32 bf16) * B(32x16 bf16) + C
  return __builtin_amdgcn_wmma_f32_16x16x32_bf16(false, a, false, b, (short)0, c,
                                                 false, false);
}

// A-fragment (16x32) from a row-major matrix with leading dim ld.
// lane m in [0,15]: row m, K = k0+0..7 and k0+16..23
// lane m+16      : row m, K = k0+8..15 and k0+24..31
static __device__ __forceinline__ v16bf load_frag_a(const bf16_t* __restrict__ base,
                                                    int ld, int row0, int k0, int lane) {
  int r  = row0 + (lane & 15);
  int kk = k0 + ((lane >> 4) << 3);
  const bf16_t* p = base + (size_t)r * ld + kk;
  v8bf lo = *(const v8bf*)(p);
  v8bf hi = *(const v8bf*)(p + 16);
  v16bf o;
#pragma unroll
  for (int i = 0; i < 8; ++i) { o[i] = lo[i]; o[i + 8] = hi[i]; }
  return o;
}

// B-fragment (32x16) where column n of B is row (row0+n) of a row-major matrix
// (i.e. B = srcT). lane n in [0,15]: K = k0+0..7 and k0+16..23; lane n+16: K = k0+8..15 and k0+24..31.
static __device__ __forceinline__ v16bf load_frag_b(const bf16_t* __restrict__ base,
                                                    int ld, int row0, int k0, int lane) {
  int r  = row0 + (lane & 15);
  int kk = k0 + ((lane >> 4) << 3);
  const bf16_t* p = base + (size_t)r * ld + kk;
  v8bf lo = *(const v8bf*)(p);
  v8bf hi = *(const v8bf*)(p + 16);
  v16bf o;
#pragma unroll
  for (int i = 0; i < 8; ++i) { o[i] = lo[i]; o[i + 8] = hi[i]; }
  return o;
}

__global__ void cvt_f32_to_bf16(const float* __restrict__ in, bf16_t* __restrict__ out,
                                int n) {
  int i = blockIdx.x * blockDim.x + threadIdx.x;
  if (i < n) out[i] = (bf16_t)in[i];
}

// y = X(4096x1024) @ W(1024x1024)^T + bias, bf16 WMMA, f32 accumulate.
// All 8 waves of a block share one 16-row X panel: staged once via TDM into
// LDS (32KB), A-fragments from ds_load_b128. Each wave owns a 16x64 N-span.
// mode 0: fused per-head L2 row-norm, store bf16 head-major [B,H,S,DK]
// mode 1: no norm, store bf16 transposed per head [B,H,DK,S]
// mode 2: no norm, store fp32 row-major [4096,1024]
__global__ __launch_bounds__(256)
void gemm_bf16_wmma(const bf16_t* __restrict__ X, const bf16_t* __restrict__ W,
                    const float* __restrict__ bias, void* __restrict__ outp, int mode) {
  __shared__ __align__(16) bf16_t sx[16 * DMODEL];   // 32KB X row-panel
  int lane = threadIdx.x & 31;
  int wave = threadIdx.x >> 5;
  int row0 = (blockIdx.x >> 1) * 16;              // block-uniform row panel
  int n0   = (((blockIdx.x & 1) << 3) + wave) * 64;  // per-wave 64-wide N span

  if (wave == 0) {
    tdm_load_2d(X + (size_t)row0 * DMODEL, sx, DMODEL, 16, DMODEL);
    TDM_WAIT0();
  }
  __syncthreads();

  v8f acc[4] = {};
  for (int k0 = 0; k0 < DMODEL; k0 += 32) {
    v16bf a = load_frag_a(sx, DMODEL, 0, k0, lane);   // LDS
#pragma unroll
    for (int t = 0; t < 4; ++t) {
      v16bf b = load_frag_b(W, DMODEL, n0 + 16 * t, k0, lane);
      acc[t] = wmma_bf16(a, b, acc[t]);
    }
    if (k0 + 64 < DMODEL)
      __builtin_prefetch(W + (size_t)(n0 + (lane & 15)) * DMODEL + k0 + 64, 0, 1);
  }

  int cl = lane & 15;
  int rb = (lane >> 4) << 3;   // rows 0..7 or 8..15 of the tile
#pragma unroll
  for (int t = 0; t < 4; ++t) {
    float bv = bias[n0 + 16 * t + cl];
#pragma unroll
    for (int j = 0; j < 8; ++j) acc[t][j] += bv;
  }

  if (mode == 0) {
    // L2 row-norm across the 64 head columns: 4 local cols + 16-lane xor tree
    float sc[8];
#pragma unroll
    for (int j = 0; j < 8; ++j) {
      float s = 0.f;
#pragma unroll
      for (int t = 0; t < 4; ++t) { float v = acc[t][j]; s += v * v; }
#pragma unroll
      for (int off = 1; off < 16; off <<= 1) s += __shfl_xor(s, off, 32);
      sc[j] = rsqrtf(s + EPSR);
    }
    bf16_t* out = (bf16_t*)outp;
#pragma unroll
    for (int t = 0; t < 4; ++t) {
      int col = n0 + 16 * t + cl;
      int h = col >> 6, d = col & 63;
#pragma unroll
      for (int j = 0; j < 8; ++j) {
        int row = row0 + rb + j;
        int b = row >> 11, s = row & (SEQ - 1);
        out[(((size_t)(b * NHEADS + h)) * SEQ + s) * DK + d] =
            (bf16_t)(acc[t][j] * sc[j]);
      }
    }
  } else if (mode == 1) {
    bf16_t* out = (bf16_t*)outp;
#pragma unroll
    for (int t = 0; t < 4; ++t) {
      int col = n0 + 16 * t + cl;
      int h = col >> 6, d = col & 63;
#pragma unroll
      for (int j = 0; j < 8; ++j) {
        int row = row0 + rb + j;
        int b = row >> 11, s = row & (SEQ - 1);
        out[(((size_t)(b * NHEADS + h)) * DK + d) * SEQ + s] = (bf16_t)acc[t][j];
      }
    }
  } else {
    float* out = (float*)outp;
#pragma unroll
    for (int t = 0; t < 4; ++t) {
      int col = n0 + 16 * t + cl;
#pragma unroll
      for (int j = 0; j < 8; ++j) {
        int row = row0 + rb + j;
        out[(size_t)row * DMODEL + col] = acc[t][j];
      }
    }
  }
}

// out[b,h,q,:] = sum_k (qn.knT)^2 * v. All 8 waves of a block share (b,h) and
// stream the same K / V^T: double-buffered 32-key chunks staged via TDM
// (wave 0 issues, s_wait_tensorcnt + barrier publishes, prefetch overlaps
// compute). Scores squared in registers, re-laid-out via wave-private LDS.
__global__ __launch_bounds__(256)
void attn_sqsim_wmma(const bf16_t* __restrict__ qn, const bf16_t* __restrict__ kn,
                     const bf16_t* __restrict__ vT, bf16_t* __restrict__ ctx) {
  __shared__ __align__(16) float  sls[8][16 * 32];   // 16KB score tiles
  __shared__ __align__(16) bf16_t skn[2][32 * DK];   // 2 x 4KB K chunks
  __shared__ __align__(16) bf16_t svt[2][DK * 32];   // 2 x 4KB V^T chunks
  int lane = threadIdx.x & 31;
  int wave = threadIdx.x >> 5;
  int bh = blockIdx.x >> 4;                 // block-uniform (b*16+h)
  int qt = ((blockIdx.x & 15) << 3) + wave; // 0..127 query tiles
  int row0 = qt * 16;
  const bf16_t* qh = qn + (size_t)bh * SEQ * DK;
  const bf16_t* kh = kn + (size_t)bh * SEQ * DK;
  const bf16_t* vh = vT + (size_t)bh * DK * SEQ;

  if (wave == 0) {
    tdm_load_2d(kh, skn[0], DK, 32, DK);        // keys 0..31, all 64 dims
    tdm_load_2d(vh, svt[0], 32, DK, SEQ);       // 64 d-rows x keys 0..31
    TDM_WAIT0();
  }

  v16bf aq0 = load_frag_a(qh, DK, row0, 0, lane);
  v16bf aq1 = load_frag_a(qh, DK, row0, 32, lane);

  v8f oacc[4] = {};
  float* sl = sls[wave];
  int cl = lane & 15;
  int rb = (lane >> 4) << 3;

  __syncthreads();
  for (int kc = 0; kc < SEQ / 32; ++kc) {
    int cur = kc & 1, nxt = cur ^ 1;
    if (wave == 0 && kc + 1 < SEQ / 32) {
      int key1 = (kc + 1) * 32;
      tdm_load_2d(kh + (size_t)key1 * DK, skn[nxt], DK, 32, DK);
      tdm_load_2d(vh + key1, svt[nxt], 32, DK, SEQ);
    }
    const bf16_t* kc_lds = skn[cur];
    const bf16_t* vc_lds = svt[cur];
#pragma unroll
    for (int nt = 0; nt < 2; ++nt) {
      v16bf b0 = load_frag_b(kc_lds, DK, 16 * nt, 0, lane);
      v16bf b1 = load_frag_b(kc_lds, DK, 16 * nt, 32, lane);
      v8f s = {};
      s = wmma_bf16(aq0, b0, s);
      s = wmma_bf16(aq1, b1, s);
      int col = 16 * nt + cl;
#pragma unroll
      for (int j = 0; j < 8; ++j) {
        float v = s[j];
        sl[(rb + j) * 32 + col] = v * v;   // squared similarity, D->LDS
      }
    }
    // reload squared scores as bf16 A-fragment (16 q-rows x 32 keys)
    v16bf a2;
    {
      int kk = (lane >> 4) << 3;
      const float* pr = sl + cl * 32 + kk;
#pragma unroll
      for (int i = 0; i < 8; ++i) {
        a2[i]     = (bf16_t)pr[i];
        a2[i + 8] = (bf16_t)pr[i + 16];
      }
    }
#pragma unroll
    for (int t = 0; t < 4; ++t) {
      v16bf bv = load_frag_b(vc_lds, 32, 16 * t, 0, lane);
      oacc[t] = wmma_bf16(a2, bv, oacc[t]);
    }
    __syncthreads();                       // all waves done with cur buffers
    if (wave == 0 && kc + 1 < SEQ / 32) TDM_WAIT0();
    __syncthreads();                       // nxt buffers now visible
  }

  int h = bh & 15, b = bh >> 4;
#pragma unroll
  for (int t = 0; t < 4; ++t) {
    int d = 16 * t + cl;
#pragma unroll
    for (int j = 0; j < 8; ++j) {
      int srow = row0 + rb + j;
      ctx[((size_t)b * SEQ + srow) * DMODEL + h * DK + d] = (bf16_t)oacc[t][j];
    }
  }
}

extern "C" void kernel_launch(void* const* d_in, const int* in_sizes, int n_in,
                              void* d_out, int out_size, void* d_ws, size_t ws_size,
                              hipStream_t stream) {
  const float* Q   = (const float*)d_in[0];
  const float* K   = (const float*)d_in[1];
  const float* V   = (const float*)d_in[2];
  const float* Wq  = (const float*)d_in[3];
  const float* Wqb = (const float*)d_in[4];
  const float* Wk  = (const float*)d_in[5];
  const float* Wkb = (const float*)d_in[6];
  const float* Wv  = (const float*)d_in[7];
  const float* Wvb = (const float*)d_in[8];
  const float* Wo  = (const float*)d_in[9];
  const float* Wob = (const float*)d_in[10];

  char* ws = (char*)d_ws;
  const size_t MB = 1ull << 20;
  bf16_t* Xq  = (bf16_t*)(ws +  0 * MB);  // 8MB each activation
  bf16_t* Xk  = (bf16_t*)(ws +  8 * MB);
  bf16_t* Xv  = (bf16_t*)(ws + 16 * MB);
  bf16_t* Wqh = (bf16_t*)(ws + 24 * MB);  // 2MB each weight
  bf16_t* Wkh = (bf16_t*)(ws + 26 * MB);
  bf16_t* Wvh = (bf16_t*)(ws + 28 * MB);
  bf16_t* Woh = (bf16_t*)(ws + 30 * MB);
  bf16_t* qn  = (bf16_t*)(ws + 32 * MB);  // [B,H,S,DK]
  bf16_t* kn  = (bf16_t*)(ws + 40 * MB);  // [B,H,S,DK]
  bf16_t* vT  = (bf16_t*)(ws + 48 * MB);  // [B,H,DK,S]
  bf16_t* ctx = Xq;  // Xq is dead after the Q projection; reuse for attn output

  int nact = MROWS * DMODEL;   // 4,194,304
  int nw   = DMODEL * DMODEL;  // 1,048,576
  cvt_f32_to_bf16<<<(nact + 255) / 256, 256, 0, stream>>>(Q, Xq, nact);
  cvt_f32_to_bf16<<<(nact + 255) / 256, 256, 0, stream>>>(K, Xk, nact);
  cvt_f32_to_bf16<<<(nact + 255) / 256, 256, 0, stream>>>(V, Xv, nact);
  cvt_f32_to_bf16<<<(nw + 255) / 256, 256, 0, stream>>>(Wq, Wqh, nw);
  cvt_f32_to_bf16<<<(nw + 255) / 256, 256, 0, stream>>>(Wk, Wkh, nw);
  cvt_f32_to_bf16<<<(nw + 255) / 256, 256, 0, stream>>>(Wv, Wvh, nw);
  cvt_f32_to_bf16<<<(nw + 255) / 256, 256, 0, stream>>>(Wo, Woh, nw);

  // 4096 wave-tiles per GEMM -> 512 blocks x 8 waves
  gemm_bf16_wmma<<<512, 256, 0, stream>>>(Xq, Wqh, Wqb, qn, 0);
  gemm_bf16_wmma<<<512, 256, 0, stream>>>(Xk, Wkh, Wkb, kn, 0);
  gemm_bf16_wmma<<<512, 256, 0, stream>>>(Xv, Wvh, Wvb, vT, 1);
  attn_sqsim_wmma<<<512, 256, 0, stream>>>(qn, kn, vT, ctx);
  gemm_bf16_wmma<<<512, 256, 0, stream>>>(ctx, Woh, Wob, d_out, 2);

  (void)in_sizes; (void)n_in; (void)out_size; (void)ws_size;
}
